// Block_89696097010240
// MI455X (gfx1250) — matrix operand order repaired and checked by reading, not compiled, not run on hardware
//
#include <hip/hip_runtime.h>
#include <math.h>

// ---------------------------------------------------------------------------
// Types / helpers
// ---------------------------------------------------------------------------
typedef __bf16 v16bf __attribute__((ext_vector_type(16)));
typedef float  v8f   __attribute__((ext_vector_type(8)));
typedef unsigned short u16;
typedef unsigned int   u32;
typedef unsigned long long u64;

union Frag {                 // one WMMA 16-bit A/B fragment: 16 bf16 = 32 bytes
    v16bf v;
    uint4 u[2];
};

__device__ __forceinline__ u16 f2bf(float f) {
    union { float f; unsigned u; } x;
    x.f = f;
    unsigned r = x.u + 0x7FFFu + ((x.u >> 16) & 1u);   // round-to-nearest-even
    return (u16)(r >> 16);
}

__device__ __forceinline__ float gelu_exact(float x) {
    return 0.5f * x * (1.0f + erff(x * 0.70710678118654752f));
}

#define WMMA_BF16(A, B, C) \
    __builtin_amdgcn_wmma_f32_16x16x32_bf16(false, (A), false, (B), (short)0, (C), false, false)

// --- DPP 16-lane butterfly reductions (no LDS traffic) ----------------------
template <int CTRL>
__device__ __forceinline__ float dpp_movf(float x) {
    int r = __builtin_amdgcn_update_dpp(0, __builtin_bit_cast(int, x),
                                        CTRL, 0xf, 0xf, true);
    return __builtin_bit_cast(float, r);
}
__device__ __forceinline__ float red16_max(float x) {
    x = fmaxf(x, dpp_movf<0xB1>(x));    // quad_perm(1,0,3,2)  xor1
    x = fmaxf(x, dpp_movf<0x4E>(x));    // quad_perm(2,3,0,1)  xor2
    x = fmaxf(x, dpp_movf<0x141>(x));   // row_half_mirror     xor7
    x = fmaxf(x, dpp_movf<0x140>(x));   // row_mirror          xor15
    return x;
}
__device__ __forceinline__ float red16_sum(float x) {
    x += dpp_movf<0xB1>(x);
    x += dpp_movf<0x4E>(x);
    x += dpp_movf<0x141>(x);
    x += dpp_movf<0x140>(x);
    return x;
}

// --- async global -> LDS (CDNA5 ASYNCcnt path) ------------------------------
__device__ __forceinline__ void async_ld_b128(u32 lds_addr, const void* gptr) {
    asm volatile("global_load_async_to_lds_b128 %0, %1, off"
                 :: "v"(lds_addr), "v"((u64)(uintptr_t)gptr)
                 : "memory");
}
#define WAIT_ASYNC(n) asm volatile("s_wait_asynccnt " #n ::: "memory")

// ---------------------------------------------------------------------------
// Problem constants
// ---------------------------------------------------------------------------
static constexpr int DIM    = 768;
static constexpr int HEADS  = 12;
static constexpr int DHEAD  = 64;
static constexpr int HIDDEN = 3072;
static constexpr int SEQ    = 1024;
static constexpr int BATCH  = 8;
static constexpr int NTOK   = BATCH * SEQ;      // 8192
static constexpr int QKVN   = 3 * DIM;          // 2304

// ---------------------------------------------------------------------------
// Weight convert + transpose: W[K][N] fp32 -> Wt[N][K] bf16
// ---------------------------------------------------------------------------
__global__ __launch_bounds__(256) void cvt_transpose_kernel(
        const float* __restrict__ W, u16* __restrict__ Wt, int K, int N) {
    size_t idx = (size_t)blockIdx.x * 256 + threadIdx.x;
    if (idx >= (size_t)K * N) return;
    int n = (int)(idx / K);
    int k = (int)(idx % K);
    Wt[idx] = f2bf(W[(size_t)k * N + n]);
}

// ---------------------------------------------------------------------------
// LayerNorm: fp32 [rows,768] -> bf16 [rows,768]
// ---------------------------------------------------------------------------
__global__ __launch_bounds__(256) void layernorm_kernel(
        const float* __restrict__ x, const float* __restrict__ g,
        const float* __restrict__ b, u16* __restrict__ out) {
    const int row = blockIdx.x;
    const int tid = threadIdx.x;
    const float* xr = x + (size_t)row * DIM;
    __shared__ float red[256];

    float v[3];
    float s = 0.0f;
#pragma unroll
    for (int i = 0; i < 3; i++) { v[i] = xr[tid + 256 * i]; s += v[i]; }
    red[tid] = s;
    __syncthreads();
    for (int off = 128; off > 0; off >>= 1) {
        if (tid < off) red[tid] += red[tid + off];
        __syncthreads();
    }
    const float mu = red[0] * (1.0f / DIM);
    __syncthreads();

    float ss = 0.0f;
#pragma unroll
    for (int i = 0; i < 3; i++) { float d = v[i] - mu; ss += d * d; }
    red[tid] = ss;
    __syncthreads();
    for (int off = 128; off > 0; off >>= 1) {
        if (tid < off) red[tid] += red[tid + off];
        __syncthreads();
    }
    const float rstd = rsqrtf(red[0] * (1.0f / DIM) + 1e-5f);

    u16* orow = out + (size_t)row * DIM;
#pragma unroll
    for (int i = 0; i < 3; i++) {
        int c = tid + 256 * i;
        orow[c] = f2bf((v[i] - mu) * rstd * g[c] + b[c]);
    }
}

// ---------------------------------------------------------------------------
// WMMA GEMM: C[M,N] = A_bf16[M,K] @ Wt_bf16[N,K]^T  (+ epilogue)
//   block = 256 threads (8 waves, 4 in M x 2 in N), tile 128x64, K-step 32
//   double-buffered async global->LDS staging (ASYNCcnt)
// ---------------------------------------------------------------------------
template <int EPI>
__global__ __launch_bounds__(256, 1) void gemm_wmma_kernel(
                                 const u16* __restrict__ A,
                                 const u16* __restrict__ Bt,
                                 const float* __restrict__ bias,
                                 const float* __restrict__ res,
                                 float* __restrict__ outF,
                                 u16* __restrict__ outB0,
                                 u16* __restrict__ outB1,
                                 u16* __restrict__ outB2,
                                 int Ntot, int Ktot) {
    __shared__ __align__(16) u16 As[2][128 * 32];
    __shared__ __align__(16) u16 Bs[2][64 * 32];

    const int tid = threadIdx.x;
    const int wid = tid >> 5;
    const int ln  = tid & 31;
    const int waveM = wid & 3;
    const int waveN = wid >> 2;
    const int g  = ln >> 4;
    const int lm = ln & 15;

    const int m0 = blockIdx.y * 128;
    const int n0 = blockIdx.x * 64;

    const int rA0 = (tid + 0)   >> 2, cA0 = ((tid + 0)   & 3) * 8;
    const int rA1 = (tid + 256) >> 2, cA1 = ((tid + 256) & 3) * 8;
    const int rB  = tid >> 2,         cB  = (tid & 3) * 8;

    auto issue_tile = [&](int k0, int buf) {
        async_ld_b128((u32)(uintptr_t)&As[buf][rA0 * 32 + cA0],
                      &A[(size_t)(m0 + rA0) * Ktot + k0 + cA0]);
        async_ld_b128((u32)(uintptr_t)&As[buf][rA1 * 32 + cA1],
                      &A[(size_t)(m0 + rA1) * Ktot + k0 + cA1]);
        async_ld_b128((u32)(uintptr_t)&Bs[buf][rB * 32 + cB],
                      &Bt[(size_t)(n0 + rB) * Ktot + k0 + cB]);
    };

    v8f acc[2][2] = {};

    const int T = Ktot >> 5;
    issue_tile(0, 0);
    for (int it = 0; it < T; ++it) {
        if (it + 1 < T) {
            issue_tile((it + 1) << 5, (it + 1) & 1);
            WAIT_ASYNC(0x3);             // previous tile's 3 asyncs complete
        } else {
            WAIT_ASYNC(0x0);
        }
        __syncthreads();

        const int buf = it & 1;
        Frag af[2], bfr[2];
#pragma unroll
        for (int i = 0; i < 2; i++) {
            int row = waveM * 32 + i * 16 + lm;
            af[i].u[0] = *(const uint4*)&As[buf][row * 32 + 8 * g];
            af[i].u[1] = *(const uint4*)&As[buf][row * 32 + 16 + 8 * g];
        }
#pragma unroll
        for (int j = 0; j < 2; j++) {
            int col = waveN * 32 + j * 16 + lm;
            bfr[j].u[0] = *(const uint4*)&Bs[buf][col * 32 + 8 * g];
            bfr[j].u[1] = *(const uint4*)&Bs[buf][col * 32 + 16 + 8 * g];
        }
#pragma unroll
        for (int i = 0; i < 2; i++)
#pragma unroll
            for (int j = 0; j < 2; j++)
                acc[i][j] = WMMA_BF16(af[i].v, bfr[j].v, acc[i][j]);
        __syncthreads();
    }

#pragma unroll
    for (int i = 0; i < 2; i++) {
#pragma unroll
        for (int j = 0; j < 2; j++) {
#pragma unroll
            for (int r = 0; r < 8; r++) {
                const int m = m0 + waveM * 32 + i * 16 + r + 8 * g;
                const int n = n0 + waveN * 32 + j * 16 + lm;
                float val = acc[i][j][r] + bias[n];
                if (EPI == 0) {
                    const int sec = n / DIM;          // 0=q 1=k 2=v
                    const int f   = n % DIM;
                    const int h   = f / DHEAD;
                    const int d   = f % DHEAD;
                    const int bh  = (m / SEQ) * HEADS + h;
                    const int tok = m % SEQ;
                    if (sec == 0)
                        outB0[((size_t)bh * SEQ + tok) * DHEAD + d] = f2bf(val * 0.125f);
                    else if (sec == 1)
                        outB1[((size_t)bh * SEQ + tok) * DHEAD + d] = f2bf(val);
                    else
                        outB2[((size_t)bh * DHEAD + d) * SEQ + tok] = f2bf(val);
                } else if (EPI == 1) {
                    outF[(size_t)m * Ntot + n] = val + res[(size_t)m * Ntot + n];
                } else {
                    outB0[(size_t)m * Ntot + n] = f2bf(gelu_exact(val));
                }
            }
        }
    }
}

// ---------------------------------------------------------------------------
// Flash-style attention. Block = 4 waves x 16 query rows. K/V tiles (32 keys)
// staged in LDS via async copies, shared by all waves, double-buffered.
//   q,k : bf16 [bh][1024][64]   (q pre-scaled)
//   vt  : bf16 [bh][64][1024]
//   O   : fp32 [bh][1024][64]
// ---------------------------------------------------------------------------
__global__ __launch_bounds__(128, 1) void attention_kernel(
                                 const u16* __restrict__ q,
                                 const u16* __restrict__ k,
                                 const u16* __restrict__ vt,
                                 float* __restrict__ O) {
    const int bh  = blockIdx.y;
    const int qb  = blockIdx.x;
    const int tid = threadIdx.x;
    const int w   = tid >> 5;
    const int ln  = tid & 31;
    const int g   = ln >> 4;
    const int lm  = ln & 15;

    __shared__ __align__(16) u16 Ks[2][32 * 64];    // [key][d]
    __shared__ __align__(16) u16 Vs[2][64 * 32];    // [d][key]
    __shared__ __align__(16) u16 Pld[4][16 * 32];   // per-wave P tile

    const u16* kbase = k  + (size_t)bh * SEQ * DHEAD;
    const u16* vbase = vt + (size_t)bh * DHEAD * SEQ;

    // staging: K tile 32x64 (256 b128 chunks), V tile 64x32 (256 chunks)
    auto issue_tile = [&](int j0, int buf) {
#pragma unroll
        for (int i = 0; i < 2; i++) {
            int c  = tid + i * 128;
            int kr = c >> 3, kc = (c & 7) * 8;
            async_ld_b128((u32)(uintptr_t)&Ks[buf][kr * 64 + kc],
                          kbase + (size_t)(j0 + kr) * DHEAD + kc);
            int vr = c >> 2, vc = (c & 3) * 8;
            async_ld_b128((u32)(uintptr_t)&Vs[buf][vr * 32 + vc],
                          vbase + (size_t)vr * SEQ + j0 + vc);
        }
    };

    const int tok0 = qb * 64 + w * 16;

    // Q fragments (16x64 = two 16x32 A-frags), resident
    Frag a0, a1;
    {
        const u16* qr = q + ((size_t)bh * SEQ + tok0 + lm) * DHEAD;
        a0.u[0] = *(const uint4*)(qr + 8 * g);
        a0.u[1] = *(const uint4*)(qr + 16 + 8 * g);
        a1.u[0] = *(const uint4*)(qr + 32 + 8 * g);
        a1.u[1] = *(const uint4*)(qr + 48 + 8 * g);
    }

    v8f o[4] = {};
    float mi[8], li[8];
#pragma unroll
    for (int r = 0; r < 8; r++) { mi[r] = -1e30f; li[r] = 0.0f; }

    issue_tile(0, 0);
    for (int it = 0; it < SEQ / 32; ++it) {
        if (it + 1 < SEQ / 32) {
            issue_tile((it + 1) * 32, (it + 1) & 1);
            WAIT_ASYNC(0x4);
        } else {
            WAIT_ASYNC(0x0);
        }
        __syncthreads();
        const int buf = it & 1;

        // --- S = Q @ K^T for 32 keys (frags from LDS) ---
        Frag bk[2][2];
#pragma unroll
        for (int kh = 0; kh < 2; kh++)
#pragma unroll
            for (int dh = 0; dh < 2; dh++) {
                const u16* kr = &Ks[buf][(kh * 16 + lm) * 64 + dh * 32 + 8 * g];
                bk[kh][dh].u[0] = *(const uint4*)kr;
                bk[kh][dh].u[1] = *(const uint4*)(kr + 16);
            }
        v8f s0 = {}, s1 = {};
        s0 = WMMA_BF16(a0.v, bk[0][0].v, s0);
        s0 = WMMA_BF16(a1.v, bk[0][1].v, s0);
        s1 = WMMA_BF16(a0.v, bk[1][0].v, s1);
        s1 = WMMA_BF16(a1.v, bk[1][1].v, s1);

        // --- online softmax: 16-lane row reductions via DPP ---
#pragma unroll
        for (int r = 0; r < 8; r++) {
            float v0 = s0[r], v1 = s1[r];
            float mx = red16_max(fmaxf(v0, v1));
            const float mnew  = fmaxf(mi[r], mx);
            const float alpha = __expf(mi[r] - mnew);
            const float p0 = __expf(v0 - mnew);
            const float p1 = __expf(v1 - mnew);
            float rs = red16_sum(p0 + p1);
            li[r] = li[r] * alpha + rs;
            mi[r] = mnew;
            Pld[w][(r + 8 * g) * 32 + lm]      = f2bf(p0);
            Pld[w][(r + 8 * g) * 32 + 16 + lm] = f2bf(p1);
#pragma unroll
            for (int t = 0; t < 4; t++) o[t][r] *= alpha;
        }
        __syncthreads();

        // --- O += P @ V (P relayout C->A via LDS; V frags from LDS) ---
        Frag pa;
        pa.u[0] = *(const uint4*)&Pld[w][lm * 32 + 8 * g];
        pa.u[1] = *(const uint4*)&Pld[w][lm * 32 + 16 + 8 * g];
#pragma unroll
        for (int t = 0; t < 4; t++) {
            Frag bv;
            const u16* vr = &Vs[buf][(t * 16 + lm) * 32 + 8 * g];
            bv.u[0] = *(const uint4*)vr;
            bv.u[1] = *(const uint4*)(vr + 16);
            o[t] = WMMA_BF16(pa.v, bv.v, o[t]);
        }
        __syncthreads();
    }

    // --- write O / l ---
#pragma unroll
    for (int t = 0; t < 4; t++)
#pragma unroll
        for (int r = 0; r < 8; r++) {
            const int tok = tok0 + r + 8 * g;
            const int d   = t * 16 + lm;
            O[((size_t)bh * SEQ + tok) * DHEAD + d] = o[t][r] / li[r];
        }
}

// ---------------------------------------------------------------------------
// O [bh][tok][d] fp32 -> attnout [token][768] bf16
// ---------------------------------------------------------------------------
__global__ __launch_bounds__(256) void repack_o_kernel(
        const float* __restrict__ O, u16* __restrict__ out) {
    size_t idx = (size_t)blockIdx.x * 256 + threadIdx.x;   // over 8192*768
    int tokg = (int)(idx / DIM);
    int f    = (int)(idx % DIM);
    int h = f / DHEAD, d = f % DHEAD;
    int b = tokg / SEQ, tok = tokg % SEQ;
    out[idx] = f2bf(O[(((size_t)b * HEADS + h) * SEQ + tok) * DHEAD + d]);
}

// ---------------------------------------------------------------------------
// Host-side orchestration
// ---------------------------------------------------------------------------
extern "C" void kernel_launch(void* const* d_in, const int* in_sizes, int n_in,
                              void* d_out, int out_size, void* d_ws, size_t ws_size,
                              hipStream_t stream) {
    const float* x      = (const float*)d_in[0];
    const float* ln1_g  = (const float*)d_in[1];
    const float* ln1_b  = (const float*)d_in[2];
    const float* qkv_w  = (const float*)d_in[3];
    const float* qkv_b  = (const float*)d_in[4];
    const float* proj_w = (const float*)d_in[5];
    const float* proj_b = (const float*)d_in[6];
    const float* ln2_g  = (const float*)d_in[7];
    const float* ln2_b  = (const float*)d_in[8];
    const float* fc1_w  = (const float*)d_in[9];
    const float* fc1_b  = (const float*)d_in[10];
    const float* fc2_w  = (const float*)d_in[11];
    const float* fc2_b  = (const float*)d_in[12];
    float* out = (float*)d_out;

    char* ws = (char*)d_ws;
    size_t off = 0;
    auto alloc = [&](size_t bytes) -> char* {
        char* p = ws + off;
        off += (bytes + 255) & ~(size_t)255;
        return p;
    };
    u16* wqkv_t  = (u16*)alloc((size_t)QKVN * DIM * 2);
    u16* wproj_t = (u16*)alloc((size_t)DIM * DIM * 2);
    u16* wfc1_t  = (u16*)alloc((size_t)HIDDEN * DIM * 2);
    u16* wfc2_t  = (u16*)alloc((size_t)DIM * HIDDEN * 2);
    u16*   h1    = (u16*)alloc((size_t)NTOK * DIM * 2);
    float* x2    = (float*)alloc((size_t)NTOK * DIM * 4);
    u16*   h2    = (u16*)alloc((size_t)NTOK * DIM * 2);
    char* R = alloc((size_t)75497472 + 4096);
    u16*   qb   = (u16*)(R);
    u16*   kb   = (u16*)(R + 12582912);
    u16*   vtb  = (u16*)(R + 25165824);
    float* Obuf = (float*)(R + 37748736);
    u16*   aob  = (u16*)(R + 62914560);
    u16*   g1   = (u16*)(R);                // fc1 output aliases q/k/vt/O

    auto blocks = [](size_t n) { return (unsigned)((n + 255) / 256); };

    cvt_transpose_kernel<<<blocks((size_t)DIM * QKVN), 256, 0, stream>>>(qkv_w, wqkv_t, DIM, QKVN);
    cvt_transpose_kernel<<<blocks((size_t)DIM * DIM), 256, 0, stream>>>(proj_w, wproj_t, DIM, DIM);
    cvt_transpose_kernel<<<blocks((size_t)DIM * HIDDEN), 256, 0, stream>>>(fc1_w, wfc1_t, DIM, HIDDEN);
    cvt_transpose_kernel<<<blocks((size_t)HIDDEN * DIM), 256, 0, stream>>>(fc2_w, wfc2_t, HIDDEN, DIM);

    layernorm_kernel<<<NTOK, 256, 0, stream>>>(x, ln1_g, ln1_b, h1);

    gemm_wmma_kernel<0><<<dim3(QKVN / 64, NTOK / 128), 256, 0, stream>>>(
        h1, wqkv_t, qkv_b, nullptr, nullptr, qb, kb, vtb, QKVN, DIM);

    attention_kernel<<<dim3(SEQ / 64, BATCH * HEADS), 128, 0, stream>>>(qb, kb, vtb, Obuf);

    repack_o_kernel<<<blocks((size_t)NTOK * DIM), 256, 0, stream>>>(Obuf, aob);

    gemm_wmma_kernel<1><<<dim3(DIM / 64, NTOK / 128), 256, 0, stream>>>(
        aob, wproj_t, proj_b, x, x2, nullptr, nullptr, nullptr, DIM, DIM);

    layernorm_kernel<<<NTOK, 256, 0, stream>>>(x2, ln2_g, ln2_b, h2);

    gemm_wmma_kernel<2><<<dim3(HIDDEN / 64, NTOK / 128), 256, 0, stream>>>(
        h2, wfc1_t, fc1_b, nullptr, nullptr, g1, nullptr, nullptr, HIDDEN, DIM);

    gemm_wmma_kernel<1><<<dim3(DIM / 64, NTOK / 128), 256, 0, stream>>>(
        g1, wfc2_t, fc2_b, x2, out, nullptr, nullptr, nullptr, DIM, HIDDEN);
}
